// SimpleTransformerDecoder_68238440398945
// MI455X (gfx1250) — matrix operand order repaired and checked by reading, not compiled, not run on hardware
//
#include <hip/hip_runtime.h>
#include <hip/hip_bf16.h>
#include <math.h>

// Problem constants (match reference)
#define LNUM  4
#define BATCH 4
#define TSEQ  1024
#define HDIM  768
#define FDIM  3072
#define NHEAD 12
#define HEADD 64
#define MT    (BATCH*TSEQ)   // 4096 rows for all activation GEMMs
#define NEGV  (-10000.0f)

typedef __attribute__((ext_vector_type(16))) __bf16 v16bf;
typedef __attribute__((ext_vector_type(8)))  __bf16 v8bf;
typedef __attribute__((ext_vector_type(2)))  __bf16 v2bf;
typedef __attribute__((ext_vector_type(8)))  float  v8f;

__device__ __forceinline__ v16bf cat16(v8bf lo, v8bf hi) {
  return __builtin_shufflevector(lo, hi, 0,1,2,3,4,5,6,7,8,9,10,11,12,13,14,15);
}
__device__ __forceinline__ v8bf cvt8(const float4 a, const float4 b) {
  v8bf r;
  r[0]=(__bf16)a.x; r[1]=(__bf16)a.y; r[2]=(__bf16)a.z; r[3]=(__bf16)a.w;
  r[4]=(__bf16)b.x; r[5]=(__bf16)b.y; r[6]=(__bf16)b.z; r[7]=(__bf16)b.w;
  return r;
}
__device__ __forceinline__ float gelu_tanh(float x) {
  const float c0 = 0.7978845608028654f, c1 = 0.044715f;
  return 0.5f * x * (1.0f + tanhf(c0 * (x + c1 * x * x * x)));
}
// Low 32 bits of a generic pointer into LDS == LDS byte offset (shared aperture).
__device__ __forceinline__ unsigned lds_off32(const void* p) {
  return (unsigned)(unsigned long long)p;
}
// CDNA5 async DMA: copy 16 bytes/lane from global (saddr + per-lane offset) into LDS.
__device__ __forceinline__ void async_copy16(unsigned lds_off, unsigned goff,
                                             unsigned long long gbase) {
  asm volatile("global_load_async_to_lds_b128 %0, %1, %2"
               :: "v"(lds_off), "v"(goff), "s"(gbase) : "memory");
}
__device__ __forceinline__ void wait_async0() {
  asm volatile("s_wait_asynccnt 0" ::: "memory");
}

// ---------------------------------------------------------------------------
// Weight pre-pack: fp32 W[K][N] -> bf16 tiles [tn][kb][n_local(128)][k(32)],
// i.e. exactly the LDS B-tile image so GEMM can DMA tiles with async loads.
// One thread per bf16 pair.
// ---------------------------------------------------------------------------
__global__ __launch_bounds__(256) void pack_w(const float* __restrict__ W,
                                              __bf16* __restrict__ out,
                                              int K, int N) {
  long p = (long)blockIdx.x * 256 + threadIdx.x;
  if (p >= (long)K * N / 2) return;
  const int ktiles = K / 32;
  long tile = p / 2048;            // 128*32/2 pairs per tile
  int  q    = (int)(p % 2048);
  int  nl   = q >> 4;              // 0..127
  int  kp   = q & 15;              // k pair 0..15
  int  tn   = (int)(tile / ktiles);
  int  kb   = (int)(tile % ktiles);
  long n = (long)tn * 128 + nl;
  long k = (long)kb * 32 + kp * 2;
  v2bf v; v[0] = (__bf16)W[k * N + n]; v[1] = (__bf16)W[(k + 1) * N + n];
  *(v2bf*)&out[p * 2] = v;
}

// ---------------------------------------------------------------------------
// GEMM: C[M,N] = act( A[M,K] @ W[K,N] + bias[N] (+ res[M,N]) )
// A fp32 (converted to bf16 into LDS); W pre-packed bf16 tiles streamed into
// double-buffered LDS via global_load_async_to_lds_b128 (ASYNCcnt).
// Block tile 128x128, K-step 32, 8 waves (2x4), wave tile 64x32 -> 4x2 WMMA.
// ---------------------------------------------------------------------------
template <int ACT, int RES>
__global__ __launch_bounds__(256) void gemm_wmma(
    const float* __restrict__ A, const __bf16* __restrict__ Wp,
    const float* __restrict__ bias, const float* __restrict__ res,
    float* __restrict__ C, int N, int K)
{
  __shared__ __align__(32) __bf16 sA[128 * 32];      // [m][k]
  __shared__ __align__(32) __bf16 sB[2][128 * 32];   // [n][k], double buffered

  const int tid  = threadIdx.x;
  const int lane = tid & 31;
  const int w    = tid >> 5;
  const int half = lane >> 4;
  const int l15  = lane & 15;
  const int wm   = w >> 2;        // 0..1
  const int wn   = w & 3;         // 0..3
  const long blockM = (long)blockIdx.y * 128;
  const long blockN = (long)blockIdx.x * 128;

  v8f acc[4][2];
  v8f vz = {0.f,0.f,0.f,0.f,0.f,0.f,0.f,0.f};
#pragma unroll
  for (int mt = 0; mt < 4; mt++)
#pragma unroll
    for (int nt = 0; nt < 2; nt++) acc[mt][nt] = vz;

  const int arow = tid >> 1;            // A staging: 0..127
  const int acg  = (tid & 1) * 16;      // 0 or 16

  const int nt_k = K / 32;
  const unsigned long long gbase =
      (unsigned long long)Wp + (unsigned long long)blockIdx.x * (unsigned)nt_k * 8192ull;
  const unsigned ldsB0 = lds_off32(&sB[0][0]);
  const unsigned ldsB1 = lds_off32(&sB[1][0]);
  const unsigned toff  = (unsigned)tid * 16u;

  // prefetch B tile 0 into buffer 0 (8 KB = 2 x 16B per thread)
  async_copy16(ldsB0 + toff, toff, gbase);
  async_copy16(ldsB0 + toff + 4096u, toff + 4096u, gbase);

  for (int kb = 0; kb < nt_k; kb++) {
    {  // stage A tile 128x32, fp32 -> bf16 (overlaps with async wait below)
      const float* ga = A + (blockM + arow) * (long)K + kb * 32 + acg;
      float4 a0 = *(const float4*)(ga + 0);
      float4 a1 = *(const float4*)(ga + 4);
      float4 a2 = *(const float4*)(ga + 8);
      float4 a3 = *(const float4*)(ga + 12);
      *(v8bf*)&sA[arow * 32 + acg + 0] = cvt8(a0, a1);
      *(v8bf*)&sA[arow * 32 + acg + 8] = cvt8(a2, a3);
    }
    wait_async0();          // current B tile resident (this wave's portion)
    __syncthreads();        // all portions resident + A staged
    if (kb + 1 < nt_k) {    // stream next B tile into the other buffer
      unsigned dst = ((kb + 1) & 1) ? ldsB1 : ldsB0;
      unsigned src = (unsigned)(kb + 1) * 8192u + toff;
      async_copy16(dst + toff, src, gbase);
      async_copy16(dst + toff + 4096u, src + 4096u, gbase);
    }
    const __bf16* sBc = sB[kb & 1];

    v16bf af[4], bfr[2];
#pragma unroll
    for (int mt = 0; mt < 4; mt++) {
      int r = wm * 64 + mt * 16 + l15;
      v8bf lo = *(const v8bf*)&sA[r * 32 + 16 * half];
      v8bf hi = *(const v8bf*)&sA[r * 32 + 16 * half + 8];
      af[mt] = cat16(lo, hi);
    }
#pragma unroll
    for (int nt = 0; nt < 2; nt++) {
      int n = wn * 32 + nt * 16 + l15;
      bfr[nt] = *(const v16bf*)&sBc[n * 32 + 16 * half];
    }
#pragma unroll
    for (int mt = 0; mt < 4; mt++)
#pragma unroll
      for (int nt = 0; nt < 2; nt++)
        acc[mt][nt] = __builtin_amdgcn_wmma_f32_16x16x32_bf16(
            false, af[mt], false, bfr[nt], (short)0, acc[mt][nt], false, false);
    __syncthreads();        // reads done before next iteration overwrites LDS
  }

  // epilogue: bias (+ residual) (+ GELU) -- fully branch-free per template
#pragma unroll
  for (int mt = 0; mt < 4; mt++) {
#pragma unroll
    for (int nt = 0; nt < 2; nt++) {
      long n = blockN + wn * 32 + nt * 16 + l15;
      float bv = bias[n];
#pragma unroll
      for (int r = 0; r < 8; r++) {
        long m = blockM + wm * 64 + mt * 16 + r + 8 * half;
        float v = acc[mt][nt][r] + bv;
        if constexpr (RES) v += res[m * (long)N + n];
        if constexpr (ACT) v = gelu_tanh(v);
        C[m * (long)N + n] = v;
      }
    }
  }
}

// ---------------------------------------------------------------------------
// Flash attention per (batch, head, 64-query block). 128 threads = 4 waves,
// each wave owns 16 query rows. Scores via WMMA (Q x K^T over HD=64), online
// softmax with lane-group shuffles, P x V via WMMA.
// mask: int32 [B, T, kvlen]; adder = mask ? 0 : -10000.
// ---------------------------------------------------------------------------
__global__ __launch_bounds__(128) void attn_wmma(
    const float* __restrict__ Q, const float* __restrict__ Kb,
    const float* __restrict__ Vb, const int* __restrict__ mask,
    float* __restrict__ O, int kvlen)
{
  __shared__ __align__(32) __bf16 sK[32 * 64];      // [k][d]
  __shared__ __align__(32) __bf16 sVt[64 * 32];     // [d][k]
  __shared__ __align__(32) __bf16 sP[4][16 * 32];   // per-wave P staging

  const int b = blockIdx.z, h = blockIdx.y;
  const int tid = threadIdx.x, lane = tid & 31, w = tid >> 5;
  const int half = lane >> 4, l15 = lane & 15;
  const int qbase = blockIdx.x * 64 + w * 16;

  v16bf qa[2];
  {
    const long qrow = (long)b * TSEQ + qbase + l15;
    const float* qp = Q + qrow * HDIM + h * HEADD;
#pragma unroll
    for (int c = 0; c < 2; c++) {
      float4 a0 = *(const float4*)(qp + c * 32 + 16 * half);
      float4 a1 = *(const float4*)(qp + c * 32 + 16 * half + 4);
      float4 b0 = *(const float4*)(qp + c * 32 + 16 * half + 8);
      float4 b1 = *(const float4*)(qp + c * 32 + 16 * half + 12);
      qa[c] = cat16(cvt8(a0, a1), cvt8(b0, b1));
    }
  }

  float mi[8], li[8];
  v8f o[4];
  v8f vz = {0.f,0.f,0.f,0.f,0.f,0.f,0.f,0.f};
#pragma unroll
  for (int r = 0; r < 8; r++) { mi[r] = -1e30f; li[r] = 0.f; }
#pragma unroll
  for (int dt = 0; dt < 4; dt++) o[dt] = vz;

  const int krow = tid >> 2, kdg = (tid & 3) * 16;
  const int va = (tid >> 3) * 2, vdg = (tid & 7) * 8;

  for (int k0 = 0; k0 < kvlen; k0 += 32) {
    __syncthreads();
    {  // stage K[32][64] bf16
      const float* g = Kb + ((long)b * kvlen + k0 + krow) * HDIM + h * HEADD + kdg;
      float4 a0 = *(const float4*)(g + 0);
      float4 a1 = *(const float4*)(g + 4);
      float4 a2 = *(const float4*)(g + 8);
      float4 a3 = *(const float4*)(g + 12);
      *(v8bf*)&sK[krow * 64 + kdg + 0] = cvt8(a0, a1);
      *(v8bf*)&sK[krow * 64 + kdg + 8] = cvt8(a2, a3);
    }
    {  // stage V transposed: sVt[d][k]
      const float* g0 = Vb + ((long)b * kvlen + k0 + va) * HDIM + h * HEADD + vdg;
      const float* g1 = g0 + HDIM;
      float4 r00 = *(const float4*)(g0 + 0);
      float4 r01 = *(const float4*)(g0 + 4);
      float4 r10 = *(const float4*)(g1 + 0);
      float4 r11 = *(const float4*)(g1 + 4);
      const float r0[8] = {r00.x,r00.y,r00.z,r00.w,r01.x,r01.y,r01.z,r01.w};
      const float r1[8] = {r10.x,r10.y,r10.z,r10.w,r11.x,r11.y,r11.z,r11.w};
#pragma unroll
      for (int j = 0; j < 8; j++) {
        v2bf p; p[0] = (__bf16)r0[j]; p[1] = (__bf16)r1[j];
        *(v2bf*)&sVt[(vdg + j) * 32 + va] = p;
      }
    }
    __syncthreads();

    v8f s0 = vz, s1 = vz;
#pragma unroll
    for (int c = 0; c < 2; c++) {
      v16bf kf0 = *(const v16bf*)&sK[(l15) * 64 + c * 32 + 16 * half];
      v16bf kf1 = *(const v16bf*)&sK[(16 + l15) * 64 + c * 32 + 16 * half];
      s0 = __builtin_amdgcn_wmma_f32_16x16x32_bf16(false, qa[c], false, kf0, (short)0, s0, false, false);
      s1 = __builtin_amdgcn_wmma_f32_16x16x32_bf16(false, qa[c], false, kf1, (short)0, s1, false, false);
    }

    float p0[8], p1[8];
#pragma unroll
    for (int r = 0; r < 8; r++) {
      int q = qbase + r + 8 * half;
      long mrow = ((long)b * TSEQ + q) * (long)kvlen + k0;
      float x0 = s0[r] * 0.125f + (mask[mrow + l15] ? 0.f : NEGV);
      float x1 = s1[r] * 0.125f + (mask[mrow + 16 + l15] ? 0.f : NEGV);
      float rm = fmaxf(x0, x1);
      rm = fmaxf(rm, __shfl_xor(rm, 1, 32));
      rm = fmaxf(rm, __shfl_xor(rm, 2, 32));
      rm = fmaxf(rm, __shfl_xor(rm, 4, 32));
      rm = fmaxf(rm, __shfl_xor(rm, 8, 32));
      float mnew = fmaxf(mi[r], rm);
      float alpha = __expf(mi[r] - mnew);
      p0[r] = __expf(x0 - mnew);
      p1[r] = __expf(x1 - mnew);
      float rs = p0[r] + p1[r];
      rs += __shfl_xor(rs, 1, 32);
      rs += __shfl_xor(rs, 2, 32);
      rs += __shfl_xor(rs, 4, 32);
      rs += __shfl_xor(rs, 8, 32);
      li[r] = li[r] * alpha + rs;
      mi[r] = mnew;
#pragma unroll
      for (int dt = 0; dt < 4; dt++) o[dt][r] *= alpha;
    }

#pragma unroll
    for (int r = 0; r < 8; r++) {
      int qq = r + 8 * half;
      sP[w][qq * 32 + l15]      = (__bf16)p0[r];
      sP[w][qq * 32 + 16 + l15] = (__bf16)p1[r];
    }
    asm volatile("s_wait_dscnt 0" ::: "memory");  // same-wave LDS RAW visibility

    v16bf pa;
    {
      v8bf lo = *(const v8bf*)&sP[w][l15 * 32 + 16 * half];
      v8bf hi = *(const v8bf*)&sP[w][l15 * 32 + 16 * half + 8];
      pa = cat16(lo, hi);
    }
#pragma unroll
    for (int dt = 0; dt < 4; dt++) {
      v16bf vf = *(const v16bf*)&sVt[(dt * 16 + l15) * 32 + 16 * half];
      o[dt] = __builtin_amdgcn_wmma_f32_16x16x32_bf16(false, pa, false, vf, (short)0, o[dt], false, false);
    }
  }

#pragma unroll
  for (int r = 0; r < 8; r++) {
    int q = qbase + r + 8 * half;
    float inv = 1.0f / li[r];
#pragma unroll
    for (int dt = 0; dt < 4; dt++)
      O[((long)b * TSEQ + q) * HDIM + h * HEADD + dt * 16 + l15] = o[dt][r] * inv;
  }
}

// ---------------------------------------------------------------------------
// LayerNorm over H=768 per row; one 256-thread block per row.
// ---------------------------------------------------------------------------
__global__ __launch_bounds__(256) void layernorm_k(
    const float* __restrict__ x, const float* __restrict__ g,
    const float* __restrict__ b, float* __restrict__ out)
{
  __shared__ float red[256];
  const long row = blockIdx.x;
  const int tid = threadIdx.x;
  const float* xr = x + row * HDIM;
  float v0 = xr[tid], v1 = xr[tid + 256], v2 = xr[tid + 512];
  red[tid] = v0 + v1 + v2;
  __syncthreads();
  for (int st = 128; st > 0; st >>= 1) {
    if (tid < st) red[tid] += red[tid + st];
    __syncthreads();
  }
  float mean = red[0] * (1.0f / HDIM);
  __syncthreads();
  float d0 = v0 - mean, d1 = v1 - mean, d2 = v2 - mean;
  red[tid] = d0 * d0 + d1 * d1 + d2 * d2;
  __syncthreads();
  for (int st = 128; st > 0; st >>= 1) {
    if (tid < st) red[tid] += red[tid + st];
    __syncthreads();
  }
  float inv = rsqrtf(red[0] * (1.0f / HDIM) + 1e-12f);
  float* orow = out + row * HDIM;
  orow[tid]       = d0 * inv * g[tid]       + b[tid];
  orow[tid + 256] = d1 * inv * g[tid + 256] + b[tid + 256];
  orow[tid + 512] = d2 * inv * g[tid + 512] + b[tid + 512];
}

// ---------------------------------------------------------------------------
// Launch: pack weights to bf16 tiles, then run the 4-layer decoder.
// Workspace: y,tmp,q,k,v,ctx (MT*H fp32 each), hbuf (MT*F fp32),
// then packed bf16 weights (8*H*H + 2*H*F per layer).  ~201 MB total.
// ---------------------------------------------------------------------------
extern "C" void kernel_launch(void* const* d_in, const int* in_sizes, int n_in,
                              void* d_out, int out_size, void* d_ws, size_t ws_size,
                              hipStream_t stream) {
  const float* tgt   = (const float*)d_in[0];
  const float* enc   = (const float*)d_in[1];
  const int*   amask = (const int*)d_in[2];
  const int*   tmask = (const int*)d_in[3];
  const float* wq_s = (const float*)d_in[4],  *bq_s = (const float*)d_in[5];
  const float* wk_s = (const float*)d_in[6],  *bk_s = (const float*)d_in[7];
  const float* wv_s = (const float*)d_in[8],  *bv_s = (const float*)d_in[9];
  const float* wo_s = (const float*)d_in[10], *bo_s = (const float*)d_in[11];
  const float* wq_c = (const float*)d_in[12], *bq_c = (const float*)d_in[13];
  const float* wk_c = (const float*)d_in[14], *bk_c = (const float*)d_in[15];
  const float* wv_c = (const float*)d_in[16], *bv_c = (const float*)d_in[17];
  const float* wo_c = (const float*)d_in[18], *bo_c = (const float*)d_in[19];
  const float* ln1g = (const float*)d_in[20], *ln1b = (const float*)d_in[21];
  const float* wi   = (const float*)d_in[22], *bi   = (const float*)d_in[23];
  const float* wf   = (const float*)d_in[24], *bf_  = (const float*)d_in[25];
  const float* ln2g = (const float*)d_in[26], *ln2b = (const float*)d_in[27];

  const long HH = (long)HDIM * HDIM, HF = (long)HDIM * FDIM;
  float* ws   = (float*)d_ws;
  float* y    = ws;
  float* tmp  = y    + (long)MT * HDIM;
  float* q    = tmp  + (long)MT * HDIM;
  float* kbuf = q    + (long)MT * HDIM;
  float* vbuf = kbuf + (long)MT * HDIM;
  float* ctx  = vbuf + (long)MT * HDIM;
  float* hbuf = ctx  + (long)MT * HDIM;
  __bf16* pw  = (__bf16*)(hbuf + (long)MT * FDIM);
  const long PWL = 8 * HH + 2 * HF;    // packed bf16 elems per layer

  hipMemcpyAsync(y, tgt, (size_t)MT * HDIM * sizeof(float),
                 hipMemcpyDeviceToDevice, stream);

  auto pack = [&](const float* W, __bf16* out, int K, int N) {
    long pairs = (long)K * N / 2;
    pack_w<<<dim3((unsigned)((pairs + 255) / 256)), 256, 0, stream>>>(W, out, K, N);
  };
  // pack all weights: per layer [qs,ks,vs,os,qc,kc,vc,oc](HH) + wi(HF) + wf(FH)
  for (int l = 0; l < LNUM; l++) {
    __bf16* base = pw + l * PWL;
    pack(wq_s + l * HH, base + 0 * HH, HDIM, HDIM);
    pack(wk_s + l * HH, base + 1 * HH, HDIM, HDIM);
    pack(wv_s + l * HH, base + 2 * HH, HDIM, HDIM);
    pack(wo_s + l * HH, base + 3 * HH, HDIM, HDIM);
    pack(wq_c + l * HH, base + 4 * HH, HDIM, HDIM);
    pack(wk_c + l * HH, base + 5 * HH, HDIM, HDIM);
    pack(wv_c + l * HH, base + 6 * HH, HDIM, HDIM);
    pack(wo_c + l * HH, base + 7 * HH, HDIM, HDIM);
    pack(wi   + l * HF, base + 8 * HH, HDIM, FDIM);          // K=H, N=F
    pack(wf   + l * HF, base + 8 * HH + HF, FDIM, HDIM);     // K=F, N=H
  }

  auto gemm = [&](const float* A, const __bf16* Wp, const float* bias,
                  const float* res, float* C, int N, int K, int act) {
    dim3 grid(N / 128, MT / 128);
    if (act)          gemm_wmma<1, 0><<<grid, 256, 0, stream>>>(A, Wp, bias, nullptr, C, N, K);
    else if (res)     gemm_wmma<0, 1><<<grid, 256, 0, stream>>>(A, Wp, bias, res, C, N, K);
    else              gemm_wmma<0, 0><<<grid, 256, 0, stream>>>(A, Wp, bias, nullptr, C, N, K);
  };
  dim3 agrid(TSEQ / 64, NHEAD, BATCH);

  for (int l = 0; l < LNUM; l++) {
    const __bf16* base = pw + l * PWL;
    // ---- self attention ----
    gemm(y, base + 0 * HH, bq_s + l * HDIM, nullptr, q,    HDIM, HDIM, 0);
    gemm(y, base + 1 * HH, bk_s + l * HDIM, nullptr, kbuf, HDIM, HDIM, 0);
    gemm(y, base + 2 * HH, bv_s + l * HDIM, nullptr, vbuf, HDIM, HDIM, 0);
    attn_wmma<<<agrid, 128, 0, stream>>>(q, kbuf, vbuf, tmask, ctx, TSEQ);
    gemm(ctx, base + 3 * HH, bo_s + l * HDIM, y, tmp, HDIM, HDIM, 0);
    layernorm_k<<<MT, 256, 0, stream>>>(tmp, ln1g + l * HDIM, ln1b + l * HDIM, y);
    // ---- cross attention (K/V from encoder) ----
    gemm(y,   base + 4 * HH, bq_c + l * HDIM, nullptr, q,    HDIM, HDIM, 0);
    gemm(enc, base + 5 * HH, bk_c + l * HDIM, nullptr, kbuf, HDIM, HDIM, 0);
    gemm(enc, base + 6 * HH, bv_c + l * HDIM, nullptr, vbuf, HDIM, HDIM, 0);
    attn_wmma<<<agrid, 128, 0, stream>>>(q, kbuf, vbuf, amask, ctx, TSEQ);
    gemm(ctx, base + 7 * HH, bo_c + l * HDIM, y, tmp, HDIM, HDIM, 0);
    layernorm_k<<<MT, 256, 0, stream>>>(tmp, ln1g + l * HDIM, ln1b + l * HDIM, y);
    // ---- FFN ----
    gemm(y, base + 8 * HH, bi + l * FDIM, nullptr, hbuf, FDIM, HDIM, 1);    // GELU
    gemm(hbuf, base + 8 * HH + HF, bf_ + l * HDIM, y, tmp, HDIM, FDIM, 0);
    float* lnout = (l == LNUM - 1) ? (float*)d_out : y;
    layernorm_k<<<MT, 256, 0, stream>>>(tmp, ln2g + l * HDIM, ln2b + l * HDIM, lnout);
  }
}
